// GraphTransformer_47768626266203
// MI455X (gfx1250) — matrix operand order repaired and checked by reading, not compiled, not run on hardware
//
#include <hip/hip_runtime.h>
#include <hip/hip_bf16.h>

#define BB 8
#define RR 1024
#define DD 256
#define HH 8
#define HDD 32
#define LL 2
#define EE 32768
#define FFD 1024
#define NEGV -1000000000.0f

typedef __attribute__((ext_vector_type(16))) __bf16 v16bf;
typedef __attribute__((ext_vector_type(8)))  __bf16 v8bf;
typedef __attribute__((ext_vector_type(8)))  float  v8f;

// ---------------- WMMA fragment helpers (16x16x32 bf16; this file's contiguous k order, k = 16*half + s) ----------------
// A (16x32 bf16): lane L<16 holds row M=L, K = 0..15 (2/vgpr);
// lanes 16..31 hold the same rows with K = 16..31.  -> two 16B loads per lane.
__device__ __forceinline__ v16bf load_a_frag(const __bf16* __restrict__ arow, int kb, int hf) {
  v8bf lo = *(const v8bf*)(arow + kb + hf * 16);
  v8bf hi = *(const v8bf*)(arow + kb + hf * 16 + 8);
  v16bf a;
#pragma unroll
  for (int j = 0; j < 8; ++j) { a[j] = lo[j]; a[8 + j] = hi[j]; }
  return a;
}
// B (32x16 bf16): lane = column n; lanes 0..15 hold K=0..15 (2/vgpr), lanes 16..31 K=16..31.
// With B[k][n] = W[n][k] (row-major weight rows), this is one contiguous 32B load.
__device__ __forceinline__ v16bf load_b_frag(const __bf16* __restrict__ wrow, int kb, int hf) {
  return *(const v16bf*)(wrow + kb + hf * 16);
}
__device__ __forceinline__ v8f wmma_bf16(v16bf a, v16bf b, v8f c) {
  return __builtin_amdgcn_wmma_f32_16x16x32_bf16(false, a, false, b, (short)0, c, false, false);
}
__device__ __forceinline__ float gelu_exact(float x) {
  return 0.5f * x * (1.0f + erff(x * 0.70710678118654752f));
}

// ---------------- mask construction ----------------
__global__ void fill_mask_kernel(float* __restrict__ mask) {
  size_t i = (size_t)blockIdx.x * blockDim.x + threadIdx.x;   // over B*R*R
  int rr = (int)(i & (size_t)(RR * RR - 1));
  int sr = rr >> 10, dr = rr & (RR - 1);
  mask[i] = (sr == dr) ? 0.0f : NEGV;
}

__global__ void edge_zero_kernel(float* __restrict__ mask, const int* __restrict__ eidx,
                                 const unsigned char* __restrict__ keep) {
  int i = blockIdx.x * blockDim.x + threadIdx.x;              // over B*E
  if (!keep[i]) return;
  int b = i >> 15, e = i & (EE - 1);
  int src = eidx[(size_t)b * 2 * EE + e];
  int dst = eidx[(size_t)b * 2 * EE + EE + e];
  mask[((size_t)b * RR + src) * RR + dst] = 0.0f;             // max(NEG,0) == store 0
}

__global__ void edge_bias_kernel(float* __restrict__ mask, const int* __restrict__ eidx,
                                 const int* __restrict__ relid,
                                 const unsigned char* __restrict__ keep,
                                 const float* __restrict__ rel_bias) {
  int i = blockIdx.x * blockDim.x + threadIdx.x;
  if (!keep[i]) return;
  int b = i >> 15, e = i & (EE - 1);
  int src = eidx[(size_t)b * 2 * EE + e];
  int dst = eidx[(size_t)b * 2 * EE + EE + e];
  atomicAdd(&mask[((size_t)b * RR + src) * RR + dst], rel_bias[relid[i]]);
}

// ---------------- dtype conversion ----------------
__global__ void f2bf_kernel(const float* __restrict__ in, __bf16* __restrict__ out, int n) {
  int i = blockIdx.x * blockDim.x + threadIdx.x;
  if (i < n) out[i] = (__bf16)in[i];
}
__global__ void init_x_kernel(const float* __restrict__ in, float* __restrict__ xf,
                              __bf16* __restrict__ xb, int n) {
  int i = blockIdx.x * blockDim.x + threadIdx.x;
  if (i < n) { float v = in[i]; xf[i] = v; xb[i] = (__bf16)v; }
}

// ---------------- generic GEMM: out = A(bf16, MxK) @ W(bf16, NxK)^T + bias ----------------
// grid.x = M/16, grid.y = N/128, block = 128 (4 waves, each wave -> 16x32 strip)
template <int MODE>  // 0: fp32 store, 1: gelu -> bf16 store
__global__ void gemm_bf16_kernel(const __bf16* __restrict__ A, const __bf16* __restrict__ W,
                                 const float* __restrict__ bias, void* __restrict__ outv,
                                 int N, int K) {
  const int wave = threadIdx.x >> 5, lane = threadIdx.x & 31;
  const int hf = lane >> 4, lm = lane & 15;
  const int m0 = blockIdx.x * 16;
  const int n0 = blockIdx.y * 128 + wave * 32;
  const __bf16* arow = A + (size_t)(m0 + lm) * K;
  const __bf16* w0 = W + (size_t)(n0 + lm) * K;
  const __bf16* w1 = W + (size_t)(n0 + 16 + lm) * K;
  v8f acc0 = {}, acc1 = {};
  for (int kb = 0; kb < K; kb += 32) {
    v16bf a = load_a_frag(arow, kb, hf);
    acc0 = wmma_bf16(a, load_b_frag(w0, kb, hf), acc0);
    acc1 = wmma_bf16(a, load_b_frag(w1, kb, hf), acc1);
  }
  const float bsa = bias[n0 + lm], bsb = bias[n0 + 16 + lm];
#pragma unroll
  for (int j = 0; j < 8; ++j) {
    int gm = m0 + hf * 8 + j;            // C/D layout: vgpr j -> row hf*8+j, col = lane&15
    float va = acc0[j] + bsa;
    float vb = acc1[j] + bsb;
    if (MODE == 0) {
      float* out = (float*)outv;
      out[(size_t)gm * N + n0 + lm] = va;
      out[(size_t)gm * N + n0 + 16 + lm] = vb;
    } else {
      __bf16* out = (__bf16*)outv;
      out[(size_t)gm * N + n0 + lm] = (__bf16)gelu_exact(va);
      out[(size_t)gm * N + n0 + 16 + lm] = (__bf16)gelu_exact(vb);
    }
  }
}

// ---------------- QKV GEMM with scatter into q (B,H,R,HD), k (B,H,R,HD), v^T (B,H,HD,R) ---------
__global__ void qkv_gemm_kernel(const __bf16* __restrict__ A, const __bf16* __restrict__ W,
                                const float* __restrict__ bias, __bf16* __restrict__ qbuf,
                                __bf16* __restrict__ kbuf, __bf16* __restrict__ vtbuf) {
  const int K = DD, wave = threadIdx.x >> 5, lane = threadIdx.x & 31;
  const int hf = lane >> 4, lm = lane & 15;
  const int m0 = blockIdx.x * 16;
  const int n0 = blockIdx.y * 128 + wave * 32;
  const __bf16* arow = A + (size_t)(m0 + lm) * K;
  const __bf16* w0 = W + (size_t)(n0 + lm) * K;
  const __bf16* w1 = W + (size_t)(n0 + 16 + lm) * K;
  v8f acc0 = {}, acc1 = {};
  for (int kb = 0; kb < K; kb += 32) {
    v16bf a = load_a_frag(arow, kb, hf);
    acc0 = wmma_bf16(a, load_b_frag(w0, kb, hf), acc0);
    acc1 = wmma_bf16(a, load_b_frag(w1, kb, hf), acc1);
  }
  const float qscale = 0.17677669529663687f;   // 1/sqrt(HD)
#pragma unroll
  for (int t2 = 0; t2 < 2; ++t2) {
    int n = n0 + t2 * 16 + lm;
    float bs = bias[n];
#pragma unroll
    for (int j = 0; j < 8; ++j) {
      float v = (t2 ? acc1[j] : acc0[j]) + bs;
      int gm = m0 + hf * 8 + j;
      int b = gm >> 10, r = gm & (RR - 1);
      if (n < DD) {
        int h = n >> 5, d = n & 31;
        qbuf[(((size_t)b * HH + h) * RR + r) * HDD + d] = (__bf16)(v * qscale);
      } else if (n < 2 * DD) {
        int n2 = n - DD, h = n2 >> 5, d = n2 & 31;
        kbuf[(((size_t)b * HH + h) * RR + r) * HDD + d] = (__bf16)v;
      } else {
        int n2 = n - 2 * DD, h = n2 >> 5, d = n2 & 31;
        vtbuf[(((size_t)b * HH + h) * HDD + d) * RR + r] = (__bf16)v;
      }
    }
  }
}

// ---------------- fused attention: one block = one (b,h,16-row q tile) vs all 1024 keys -------
// block = 128 threads (4 waves); 16x1024 fp32 score strip lives in LDS (64 KB of the WGP's 320 KB)
__global__ void attn_kernel(const __bf16* __restrict__ qbuf, const __bf16* __restrict__ kbuf,
                            const __bf16* __restrict__ vtbuf, const float* __restrict__ mask,
                            __bf16* __restrict__ attnout) {
  __shared__ float sc[16 * RR];
  const int qt = blockIdx.x, h = blockIdx.y, b = blockIdx.z;
  const int wave = threadIdx.x >> 5, lane = threadIdx.x & 31;
  const int hf = lane >> 4, lm = lane & 15;
  const size_t bh = (size_t)b * HH + h;

  // q fragment: K dim == HD == 32 -> single wmma per score tile
  const __bf16* qrow = qbuf + (bh * RR + qt * 16 + lm) * HDD;
  v16bf aq = load_a_frag(qrow, 0, hf);

  for (int kt = wave; kt < RR / 16; kt += 4) {
    v16bf bk = *(const v16bf*)(kbuf + (bh * RR + kt * 16 + lm) * HDD + hf * 16);
    v8f s = {};
    s = wmma_bf16(aq, bk, s);
    int col = kt * 16 + lm;
    const float* mrow = mask + ((size_t)b * RR + qt * 16) * RR + col;
#pragma unroll
    for (int j = 0; j < 8; ++j) {
      int row = hf * 8 + j;
      sc[row * RR + col] = s[j] + mrow[(size_t)row * RR];
    }
  }
  __syncthreads();

  // softmax: 8 threads per row, 128 cols each, shuffle-reduce within the 8-lane group
  {
    int r = threadIdx.x >> 3;
    float* srow = sc + r * RR + (threadIdx.x & 7) * 128;
    float mx = -3.4e38f;
    for (int c = 0; c < 128; ++c) mx = fmaxf(mx, srow[c]);
    for (int o = 1; o < 8; o <<= 1) mx = fmaxf(mx, __shfl_xor(mx, o, 32));
    float sum = 0.f;
    for (int c = 0; c < 128; ++c) { float e = __expf(srow[c] - mx); srow[c] = e; sum += e; }
    for (int o = 1; o < 8; o <<= 1) sum += __shfl_xor(sum, o, 32);
    float inv = 1.f / sum;
    for (int c = 0; c < 128; ++c) srow[c] *= inv;
  }
  __syncthreads();

  // attn @ V:  out tile 16x32 (= 2 col-tiles).  waves 0/1 -> j in [0,512), waves 2/3 -> [512,1024)
  const int ct = wave & 1;
  const int jlo = (wave >> 1) * 512;
  const __bf16* vrow = vtbuf + (bh * HDD + ct * 16 + lm) * RR;
  v8f acc = {};
  for (int kb = jlo; kb < jlo + 512; kb += 32) {
    v16bf aa;
    const float* srow = sc + lm * RR + kb;
#pragma unroll
    for (int j = 0; j < 8; ++j) {
      aa[j] = (__bf16)srow[hf * 16 + j];
      aa[8 + j] = (__bf16)srow[hf * 16 + 8 + j];
    }
    v16bf bv = *(const v16bf*)(vrow + kb + hf * 16);
    acc = wmma_bf16(aa, bv, acc);
  }
  __syncthreads();                       // scores no longer needed -> reuse LDS for partials
  if (wave >= 2) {
#pragma unroll
    for (int j = 0; j < 8; ++j) sc[ct * 256 + (hf * 8 + j) * 16 + lm] = acc[j];
  }
  __syncthreads();
  if (wave < 2) {
#pragma unroll
    for (int j = 0; j < 8; ++j) {
      float v = acc[j] + sc[ct * 256 + (hf * 8 + j) * 16 + lm];
      int row = qt * 16 + hf * 8 + j;
      int col = h * HDD + ct * 16 + lm;
      attnout[((size_t)b * RR + row) * DD + col] = (__bf16)v;
    }
  }
}

// ---------------- fused residual + LayerNorm (one block per row of D=256) ----------------
__global__ void residual_ln_kernel(const float* __restrict__ x, const float* __restrict__ y,
                                   const float* __restrict__ g, const float* __restrict__ bta,
                                   float* __restrict__ outf, __bf16* __restrict__ outb) {
  const int row = blockIdx.x, t = threadIdx.x;
  float v = x[(size_t)row * DD + t] + y[(size_t)row * DD + t];
  __shared__ float red[8];
  int wid = t >> 5, ln = t & 31;
  float s = v;
  for (int o = 16; o >= 1; o >>= 1) s += __shfl_xor(s, o, 32);
  if (ln == 0) red[wid] = s;
  __syncthreads();
  if (t == 0) { float a = 0; for (int i = 0; i < 8; ++i) a += red[i]; red[0] = a; }
  __syncthreads();
  float mu = red[0] * (1.0f / DD);
  float dv = v - mu;
  float s2 = dv * dv;
  for (int o = 16; o >= 1; o >>= 1) s2 += __shfl_xor(s2, o, 32);
  __syncthreads();
  if (ln == 0) red[wid] = s2;
  __syncthreads();
  if (t == 0) { float a = 0; for (int i = 0; i < 8; ++i) a += red[i]; red[0] = a; }
  __syncthreads();
  float rs = rsqrtf(red[0] * (1.0f / DD) + 1e-5f);
  float o = dv * rs * g[t] + bta[t];
  outf[(size_t)row * DD + t] = o;
  if (outb) outb[(size_t)row * DD + t] = (__bf16)o;
}

// ---------------- launch ----------------
extern "C" void kernel_launch(void* const* d_in, const int* in_sizes, int n_in,
                              void* d_out, int out_size, void* d_ws, size_t ws_size,
                              hipStream_t stream) {
  const float* node_emb = (const float*)d_in[0];
  const int* edge_index = (const int*)d_in[1];
  const int* relation_id = (const int*)d_in[2];
  const unsigned char* keep_mask = (const unsigned char*)d_in[3];
  const float* rel_bias = (const float*)d_in[4];
  const float* Wqkv = (const float*)d_in[5];
  const float* bqkv = (const float*)d_in[6];
  const float* Wo = (const float*)d_in[7];
  const float* bo = (const float*)d_in[8];
  const float* ln1_g = (const float*)d_in[9];
  const float* ln1_b = (const float*)d_in[10];
  const float* W1 = (const float*)d_in[11];
  const float* b1 = (const float*)d_in[12];
  const float* W2 = (const float*)d_in[13];
  const float* b2 = (const float*)d_in[14];
  const float* ln2_g = (const float*)d_in[15];
  const float* ln2_b = (const float*)d_in[16];

  char* w = (char*)d_ws;
  float* mask = (float*)w;    w += (size_t)BB * RR * RR * 4;        // 32 MB
  float* x = (float*)w;       w += (size_t)BB * RR * DD * 4;        // 8 MB
  float* tmp = (float*)w;     w += (size_t)BB * RR * DD * 4;        // 8 MB
  __bf16* xb = (__bf16*)w;    w += (size_t)BB * RR * DD * 2;
  __bf16* qbuf = (__bf16*)w;  w += (size_t)BB * HH * RR * HDD * 2;
  __bf16* kbuf = (__bf16*)w;  w += (size_t)BB * HH * RR * HDD * 2;
  __bf16* vtbuf = (__bf16*)w; w += (size_t)BB * HH * HDD * RR * 2;
  __bf16* attnout = (__bf16*)w; w += (size_t)BB * RR * DD * 2;
  __bf16* hbuf = (__bf16*)w;  w += (size_t)BB * RR * FFD * 2;       // 16 MB
  __bf16* Wqkv_b = (__bf16*)w; w += (size_t)LL * 3 * DD * DD * 2;
  __bf16* Wo_b = (__bf16*)w;  w += (size_t)LL * DD * DD * 2;
  __bf16* W1_b = (__bf16*)w;  w += (size_t)LL * FFD * DD * 2;
  __bf16* W2_b = (__bf16*)w;  w += (size_t)LL * DD * FFD * 2;

  // mask
  fill_mask_kernel<<<(BB * RR * RR) / 256, 256, 0, stream>>>(mask);
  edge_zero_kernel<<<(BB * EE) / 256, 256, 0, stream>>>(mask, edge_index, keep_mask);
  edge_bias_kernel<<<(BB * EE) / 256, 256, 0, stream>>>(mask, edge_index, relation_id,
                                                        keep_mask, rel_bias);
  // weight / activation conversion
  f2bf_kernel<<<(LL * 3 * DD * DD) / 256, 256, 0, stream>>>(Wqkv, Wqkv_b, LL * 3 * DD * DD);
  f2bf_kernel<<<(LL * DD * DD) / 256, 256, 0, stream>>>(Wo, Wo_b, LL * DD * DD);
  f2bf_kernel<<<(LL * FFD * DD) / 256, 256, 0, stream>>>(W1, W1_b, LL * FFD * DD);
  f2bf_kernel<<<(LL * DD * FFD) / 256, 256, 0, stream>>>(W2, W2_b, LL * DD * FFD);
  init_x_kernel<<<(BB * RR * DD) / 256, 256, 0, stream>>>(node_emb, x, xb, BB * RR * DD);

  const int M = BB * RR;  // 8192 -> 512 row tiles
  for (int l = 0; l < LL; ++l) {
    qkv_gemm_kernel<<<dim3(M / 16, (3 * DD) / 128), 128, 0, stream>>>(
        xb, Wqkv_b + (size_t)l * 3 * DD * DD, bqkv + l * 3 * DD, qbuf, kbuf, vtbuf);
    attn_kernel<<<dim3(RR / 16, HH, BB), 128, 0, stream>>>(qbuf, kbuf, vtbuf, mask, attnout);
    gemm_bf16_kernel<0><<<dim3(M / 16, DD / 128), 128, 0, stream>>>(
        attnout, Wo_b + (size_t)l * DD * DD, bo + l * DD, tmp, DD, DD);
    residual_ln_kernel<<<M, DD, 0, stream>>>(x, tmp, ln1_g + l * DD, ln1_b + l * DD, x, xb);
    gemm_bf16_kernel<1><<<dim3(M / 16, FFD / 128), 128, 0, stream>>>(
        xb, W1_b + (size_t)l * FFD * DD, b1 + l * FFD, hbuf, FFD, DD);
    gemm_bf16_kernel<0><<<dim3(M / 16, DD / 128), 128, 0, stream>>>(
        hbuf, W2_b + (size_t)l * DD * FFD, b2 + l * DD, tmp, DD, FFD);
    float* dst = (l == LL - 1) ? (float*)d_out : x;
    residual_ln_kernel<<<M, DD, 0, stream>>>(x, tmp, ln2_g + l * DD, ln2_b + l * DD, dst, xb);
  }
}
